// GraphResNetGAT_13752485282058
// MI455X (gfx1250) — matrix operand order repaired and checked by reading, not compiled, not run on hardware
//
#include <hip/hip_runtime.h>
#include <hip/hip_bf16.h>

#define CHN   128
#define HEADS 4
#define HC    32
#define LPAD  8      // bf16 elements of row padding in LDS
#define MT    80     // GEMM M-tile rows per block (50000 = 625*80)
#define SUBT  5      // 16-row sub-tiles per block

typedef __attribute__((ext_vector_type(16))) __bf16 v16bf;
typedef __attribute__((ext_vector_type(8)))  float  v8f;
typedef __attribute__((ext_vector_type(4)))  float  f4;

// ---------- monotonic float<->uint encoding for atomic segment-max ----------
__device__ __forceinline__ unsigned fenc(float f) {
  unsigned b = __float_as_uint(f);
  return b ^ (unsigned)(((int)b >> 31) | 0x80000000u);
}
__device__ __forceinline__ float fdec(unsigned u) {
  unsigned b = (u & 0x80000000u) ? (u ^ 0x80000000u) : ~u;
  return __uint_as_float(b);
}
#define ENC_NEG_INF 0x007FFFFFu   // fenc(-inf)

// ---------- W[k][n] fp32 -> Wt[n][k] bf16 (once per conv; 16K elements) ----------
__global__ void k_wcvt(const float* __restrict__ W, __bf16* __restrict__ Wt) {
  int idx = blockIdx.x * blockDim.x + threadIdx.x;   // idx = n*128 + k
  if (idx >= CHN * CHN) return;
  int n = idx >> 7, k = idx & 127;
  Wt[idx] = (__bf16)W[k * CHN + n];
}

// ---------- H = X @ W via v_wmma_f32_16x16x32_bf16 ----------
// Block: 80 rows x 128 cols. 8 waves; wave w owns cols [16w,16w+16) and 5 row
// sub-tiles, reusing each B fragment across 5 WMMAs.
__global__ __launch_bounds__(256) void k_gemm_wmma(const float* __restrict__ X,
                                                   const __bf16* __restrict__ Wt,
                                                   float* __restrict__ H, int nrows) {
  __shared__ __bf16 sW[CHN][CHN + LPAD];   // sW[n][k]
  __shared__ __bf16 sA[MT][CHN + LPAD];    // sA[m][k]
  const int tid  = threadIdx.x;
  const int row0 = blockIdx.x * MT;

  // stage pre-converted bf16 weights (32KB) as dword copies
  for (int idx = tid; idx < CHN * (CHN / 2); idx += 256) {   // 8192 dwords
    int n = idx >> 6, q = idx & 63;
    ((unsigned*)&sW[n][0])[q] = ((const unsigned*)Wt)[n * 64 + q];
  }
  // stage + convert A tile (80 x 128 fp32 -> bf16), float4 granularity
  for (int idx = tid; idx < MT * (CHN / 4); idx += 256) {    // 2560 quads
    int r = idx >> 5, c4 = (idx & 31) << 2;
    int row = row0 + r;
    if (row < nrows) {
      f4 v = *(const f4*)(X + (long)row * CHN + c4);
      sA[r][c4 + 0] = (__bf16)v[0];
      sA[r][c4 + 1] = (__bf16)v[1];
      sA[r][c4 + 2] = (__bf16)v[2];
      sA[r][c4 + 3] = (__bf16)v[3];
    }
  }
  __syncthreads();

  const int wave = tid >> 5;
  const int lane = tid & 31;
  const int half = lane >> 4;
  const int m    = lane & 15;
  const int n    = (wave << 4) + (lane & 15);

  v8f acc[SUBT] = {};
#pragma unroll
  for (int s = 0; s < 4; ++s) {
    const int k0 = s * 32;
    v16bf b;
#pragma unroll
    for (int j = 0; j < 16; ++j)            // B 32x16: contiguous K run per lane half
      b[j] = sW[n][k0 + (half << 4) + j];
#pragma unroll
    for (int t = 0; t < SUBT; ++t) {
      v16bf a;
#pragma unroll
      for (int j = 0; j < 16; ++j) {
        // A 16x32: element j holds K = half*16 + j (contiguous, matches B)
        const int ka = k0 + (half << 4) + j;
        a[j] = sA[t * 16 + m][ka];
      }
      acc[t] = __builtin_amdgcn_wmma_f32_16x16x32_bf16(false, a, false, b,
                                                       (short)0, acc[t], false, false);
    }
  }
#pragma unroll
  for (int t = 0; t < SUBT; ++t) {
#pragma unroll
    for (int r = 0; r < 8; ++r) {
      const int row = row0 + t * 16 + r + (half << 3);  // C/D layout: m = r + 8*half
      if (row < nrows) H[(long)row * CHN + n] = acc[t][r];
    }
  }
}

// ---------- ce[h] = dot(W_e[0, h*32:...], att_e[h,:])  (a_e = edge_attr * ce) ----------
__global__ void k_ce(const float* __restrict__ We, const float* __restrict__ Ae,
                     float* __restrict__ ce) {
  int t = threadIdx.x;                 // 128 threads, one wave32 per head
  float p = We[t] * Ae[t];
#pragma unroll
  for (int off = 16; off > 0; off >>= 1) p += __shfl_down(p, off, 32);
  if ((t & 31) == 0) ce[t >> 5] = p;
}

// ---------- per-node attention terms ----------
__global__ void k_node_att(const float* __restrict__ H, const float* __restrict__ as,
                           const float* __restrict__ ad, float* __restrict__ a_src,
                           float* __restrict__ a_dst, int n) {
  int idx = blockIdx.x * blockDim.x + threadIdx.x;
  if (idx >= n * HEADS) return;
  int node = idx >> 2, hh = idx & 3;
  const float* hp = H + (long)node * CHN + hh * HC;
  float s1 = 0.f, s2 = 0.f;
#pragma unroll
  for (int c = 0; c < HC; ++c) {
    float v = hp[c];
    s1 += v * as[hh * HC + c];
    s2 += v * ad[hh * HC + c];
  }
  a_src[idx] = s1;
  a_dst[idx] = s2;
}

// ---------- init: out accumulator = bias (folds "+ b"), amax=-inf, denom=0 ----------
__global__ void k_init(float* __restrict__ out, const float* __restrict__ bias,
                       unsigned* __restrict__ amax, float* __restrict__ denom, int n) {
  int idx = blockIdx.x * blockDim.x + threadIdx.x;
  if (idx < n * CHN) out[idx] = bias[idx & 127];
  if (idx < n * HEADS) { amax[idx] = ENC_NEG_INF; denom[idx] = 0.f; }
}

// ---------- edge logits + segment max (edge streams are non-temporal) ----------
__global__ void k_edge_logits(const int* __restrict__ ei, const float* __restrict__ ea,
                              const float* __restrict__ a_src, const float* __restrict__ a_dst,
                              const float* __restrict__ ce, float* __restrict__ alpha,
                              unsigned* __restrict__ amax, int ne) {
  int idx = blockIdx.x * blockDim.x + threadIdx.x;
  if (idx >= ne * HEADS) return;
  int e = idx >> 2, hh = idx & 3;
  int s = __builtin_nontemporal_load(ei + e);
  int d = __builtin_nontemporal_load(ei + ne + e);
  float av = __builtin_nontemporal_load(ea + e);
  float al = a_src[s * HEADS + hh] + a_dst[d * HEADS + hh] + av * ce[hh];
  al = al > 0.f ? al : 0.2f * al;                  // leaky_relu(0.2)
  __builtin_nontemporal_store(al, alpha + idx);
  atomicMax(&amax[d * HEADS + hh], fenc(al));
}

// ---------- exp(alpha - max) + segment sum (alpha buffer rewritten in place) ----------
__global__ void k_edge_exp(const int* __restrict__ ei, float* __restrict__ alpha,
                           const unsigned* __restrict__ amax, float* __restrict__ denom,
                           int ne) {
  int idx = blockIdx.x * blockDim.x + threadIdx.x;
  if (idx >= ne * HEADS) return;
  int e = idx >> 2, hh = idx & 3;
  int d = __builtin_nontemporal_load(ei + ne + e);
  float m = fdec(amax[d * HEADS + hh]);
  if (!(m > -3.0e38f && m < 3.0e38f)) m = 0.f;     // isfinite fix-up
  float ex = __expf(__builtin_nontemporal_load(alpha + idx) - m);
  __builtin_nontemporal_store(ex, alpha + idx);
  atomicAdd(&denom[d * HEADS + hh], ex);
}

// ---------- weighted gather/scatter aggregation: wave per edge, float4 per lane ----------
__global__ __launch_bounds__(256) void k_edge_agg(const int* __restrict__ ei,
                                                  const float* __restrict__ H,
                                                  const float* __restrict__ ex,
                                                  const float* __restrict__ denom,
                                                  float* __restrict__ out, int ne) {
  int idx  = blockIdx.x * blockDim.x + threadIdx.x;
  int e    = idx >> 5;
  int lane = idx & 31;
  if (e >= ne) return;
  int s = __builtin_nontemporal_load(ei + e);
  int d = __builtin_nontemporal_load(ei + ne + e);
  int hh = lane >> 3;                                // 4 ch/lane -> one head per lane
  float exv = __builtin_nontemporal_load(ex + e * HEADS + hh);
  float att = exv / (denom[d * HEADS + hh] + 1e-16f);
  int c0 = lane * 4;
  const f4 hv = *(const f4*)(H + (long)s * CHN + c0); // node features: keep L2-resident
  float* op = out + (long)d * CHN + c0;
  atomicAdd(op + 0, hv[0] * att);
  atomicAdd(op + 1, hv[1] * att);
  atomicAdd(op + 2, hv[2] * att);
  atomicAdd(op + 3, hv[3] * att);
}

// ---------- per-channel mean/var -> fused scale/shift ----------
__global__ __launch_bounds__(256) void k_colstats(const float* __restrict__ T,
                                                  const float* __restrict__ gamma,
                                                  const float* __restrict__ beta,
                                                  float* __restrict__ scale,
                                                  float* __restrict__ shift, int n) {
  __shared__ float ss[256], ss2[256];
  int c = blockIdx.x, t = threadIdx.x;
  float s = 0.f, s2 = 0.f;
  for (int r = t; r < n; r += 256) {
    float v = T[(long)r * CHN + c];
    s += v; s2 += v * v;
  }
  ss[t] = s; ss2[t] = s2;
  __syncthreads();
  for (int k = 128; k > 0; k >>= 1) {
    if (t < k) { ss[t] += ss[t + k]; ss2[t] += ss2[t + k]; }
    __syncthreads();
  }
  if (t == 0) {
    float mu  = ss[0] / (float)n;
    float var = ss2[0] / (float)n - mu * mu;
    float sc  = gamma[c] * rsqrtf(var + 1e-5f);
    scale[c] = sc;
    shift[c] = beta[c] - mu * sc;
  }
}

// ---------- y = relu(bn(T) [+ residual]) ----------
__global__ void k_bn_apply(const float* __restrict__ T, const float* __restrict__ scale,
                           const float* __restrict__ shift, const float* __restrict__ res,
                           float* __restrict__ O, int total) {
  int idx = blockIdx.x * blockDim.x + threadIdx.x;
  if (idx >= total) return;
  int c = idx & 127;
  float v = T[idx] * scale[c] + shift[c];
  if (res) v += res[idx];
  O[idx] = v > 0.f ? v : 0.f;
}

// ------------------------------------------------------------------
static void run_conv(const float* xin, const float* W, const float* atts,
                     const float* attd, const float* We, const float* Ae,
                     const float* bias, const int* ei, const float* ea,
                     float* H, float* T, float* ASRC, float* ADST,
                     unsigned* AMAX, float* DEN, float* ALPHA, float* CE,
                     __bf16* WT, int n, int ne, hipStream_t stream) {
  k_wcvt<<<(CHN * CHN + 255) / 256, 256, 0, stream>>>(W, WT);
  k_gemm_wmma<<<(n + MT - 1) / MT, 256, 0, stream>>>(xin, WT, H, n);
  k_ce<<<1, 128, 0, stream>>>(We, Ae, CE);
  k_node_att<<<(n * HEADS + 255) / 256, 256, 0, stream>>>(H, atts, attd, ASRC, ADST, n);
  k_init<<<(n * CHN + 255) / 256, 256, 0, stream>>>(T, bias, AMAX, DEN, n);
  k_edge_logits<<<(ne * HEADS + 255) / 256, 256, 0, stream>>>(ei, ea, ASRC, ADST, CE, ALPHA, AMAX, ne);
  k_edge_exp<<<(ne * HEADS + 255) / 256, 256, 0, stream>>>(ei, ALPHA, AMAX, DEN, ne);
  k_edge_agg<<<(ne * 32 + 255) / 256, 256, 0, stream>>>(ei, H, ALPHA, DEN, T, ne);
}

extern "C" void kernel_launch(void* const* d_in, const int* in_sizes, int n_in,
                              void* d_out, int out_size, void* d_ws, size_t ws_size,
                              hipStream_t stream) {
  const float* x  = (const float*)d_in[0];
  const int*   ei = (const int*)d_in[1];
  const float* ea = (const float*)d_in[2];
  // d_in[3] = batch (unused, single graph)

  const int n  = in_sizes[0] / CHN;   // 50000
  const int ne = in_sizes[2];         // 800000
  const long nch = (long)n * CHN;
  const long n4  = (long)n * HEADS;
  const long e4  = (long)ne * HEADS;

  float* ws    = (float*)d_ws;
  float* H     = ws;                       // [n,128] projected features
  float* T     = H + nch;                  // [n,128] conv accumulator
  float* U     = T + nch;                  // [n,128] post-bn/relu intermediate
  float* XC    = U + nch;                  // [n,128] block output carrier
  float* ASRC  = XC + nch;                 // [n,4]
  float* ADST  = ASRC + n4;                // [n,4]
  unsigned* AMAX = (unsigned*)(ADST + n4); // [n,4] encoded segment max
  float* DEN   = (float*)(AMAX + n4);      // [n,4]
  float* ALPHA = DEN + n4;                 // [e,4] logits then exp
  float* CE    = ALPHA + e4;               // [4]
  float* SCALE = CE + 8;                   // [128]
  float* SHIFT = SCALE + CHN;              // [128]
  __bf16* WT   = (__bf16*)(SHIFT + CHN);   // [128,128] bf16 transposed weights

  const float* xin = x;
  for (int blk = 0; blk < 3; ++blk) {
    const int base = 4 + 14 * blk;
    const float* g1W  = (const float*)d_in[base + 0];
    const float* g1as = (const float*)d_in[base + 1];
    const float* g1ad = (const float*)d_in[base + 2];
    const float* g1We = (const float*)d_in[base + 3];
    const float* g1Ae = (const float*)d_in[base + 4];
    const float* g1b  = (const float*)d_in[base + 5];
    const float* g2W  = (const float*)d_in[base + 6];
    const float* g2as = (const float*)d_in[base + 7];
    const float* g2ad = (const float*)d_in[base + 8];
    const float* g2We = (const float*)d_in[base + 9];
    const float* g2Ae = (const float*)d_in[base + 10];
    const float* g2b  = (const float*)d_in[base + 11];
    const float* gam  = (const float*)d_in[base + 12];
    const float* bet  = (const float*)d_in[base + 13];

    // conv1 -> bn -> relu -> U
    run_conv(xin, g1W, g1as, g1ad, g1We, g1Ae, g1b, ei, ea,
             H, T, ASRC, ADST, AMAX, DEN, ALPHA, CE, WT, n, ne, stream);
    k_colstats<<<CHN, 256, 0, stream>>>(T, gam, bet, SCALE, SHIFT, n);
    k_bn_apply<<<(int)((nch + 255) / 256), 256, 0, stream>>>(T, SCALE, SHIFT, nullptr, U, (int)nch);

    // conv2 -> bn -> +residual -> relu
    run_conv(U, g2W, g2as, g2ad, g2We, g2Ae, g2b, ei, ea,
             H, T, ASRC, ADST, AMAX, DEN, ALPHA, CE, WT, n, ne, stream);
    k_colstats<<<CHN, 256, 0, stream>>>(T, gam, bet, SCALE, SHIFT, n);
    float* outp = (blk == 2) ? (float*)d_out : XC;
    k_bn_apply<<<(int)((nch + 255) / 256), 256, 0, stream>>>(T, SCALE, SHIFT, xin, outp, (int)nch);
    xin = XC;
  }
}
